// ContextAwareGating_40235253629032
// MI455X (gfx1250) — matrix operand order repaired and checked by reading, not compiled, not run on hardware
//
#include <hip/hip_runtime.h>
#include <math.h>

typedef _Float16 half16v __attribute__((ext_vector_type(16)));
typedef _Float16 half8v  __attribute__((ext_vector_type(8)));
typedef float    float8v __attribute__((ext_vector_type(8)));

union V16 { half16v v; half8v h[2]; };

__device__ __forceinline__ float gelu_exact(float x) {
    return 0.5f * x * (1.0f + erff(x * 0.70710678118654752f));
}

// ---------------- weight pack: f32 -> f16, padded for WMMA tiles ----------------
// W1: [528][1056] (exact multiples)         -> ws[0 ..)
// W2: [272][544]  (pad from [264][528])     -> ws[557568 ..)
// W3: [16][288]   (pad from [16][264])      -> ws[705536 ..)
__global__ void pack_weights_kernel(const float* __restrict__ w1,
                                    const float* __restrict__ w2,
                                    const float* __restrict__ w3,
                                    _Float16* __restrict__ ws) {
    int idx = blockIdx.x * blockDim.x + threadIdx.x;
    const int T1 = 528 * 1056;
    const int T2 = 272 * 544;
    const int T3 = 16 * 288;
    if (idx < T1) {
        ws[idx] = (_Float16)w1[idx];
    } else if (idx < T1 + T2) {
        int j = idx - T1;
        int n = j / 544, k = j - n * 544;
        float v = (n < 264 && k < 528) ? w2[n * 528 + k] : 0.0f;
        ws[idx] = (_Float16)v;
    } else if (idx < T1 + T2 + T3) {
        int j = idx - T1 - T2;
        int n = j / 288, k = j - n * 288;
        float v = (k < 264) ? w3[n * 264 + k] : 0.0f;
        ws[idx] = (_Float16)v;
    }
}

// ---------------- fused gating kernel: 64 rows per workgroup ----------------
// LDS strides (chosen so 16 consecutive rows map to distinct bank groups):
//   fusion A (f16): stride 1096 halves  (548 dwords, 548%64=36 -> 16 distinct)
//   h buffer (f32): stride 548 floats   (h1) / 276 floats (h2)
//   A2 (f16): stride 552 halves; A3 (f16): stride 296 halves
__global__ __launch_bounds__(256)
void gating_fused_kernel(
    const float* __restrict__ x, const float* __restrict__ ctx,
    const float* __restrict__ en_g, const float* __restrict__ en_b,
    const float* __restrict__ cn_g, const float* __restrict__ cn_b,
    const float* __restrict__ cp_w1, const float* __restrict__ cp_b1,
    const float* __restrict__ cp_ln1_g, const float* __restrict__ cp_ln1_b,
    const float* __restrict__ cp_w2, const float* __restrict__ cp_b2,
    const float* __restrict__ cp_ln2_g, const float* __restrict__ cp_ln2_b,
    const float* __restrict__ cf_g, const float* __restrict__ cf_b,
    const float* __restrict__ gp_b1, const float* __restrict__ g1g, const float* __restrict__ g1b,
    const float* __restrict__ gp_b2, const float* __restrict__ g2g, const float* __restrict__ g2b,
    const float* __restrict__ gp_b3,
    const _Float16* __restrict__ Wh1, const _Float16* __restrict__ Wh2,
    const _Float16* __restrict__ Wh3,
    float* __restrict__ out, int Nrows)
{
    __shared__ _Float16 sA[64 * 1096];   // 140,288 B : fusion A / A2 / A3 (reused)
    __shared__ float    sH[64 * 548];    // 140,288 B : ctx-proj h / h1 / h2 (reused)
    __shared__ float    sLog[64 * 16];   //   4,096 B : logits
    __shared__ float    sRed[64 * 8];    //   2,048 B : per-row partial reductions

    const int tid  = threadIdx.x;
    const int row  = tid >> 2;     // 0..63
    const int sub  = tid & 3;      // 0..3
    const int wv   = tid >> 5;     // wave 0..7
    const int lane = tid & 31;
    const int lrow = lane & 15;
    const int lhi  = lane >> 4;
    const int r0   = blockIdx.x * 64;
    const int gr   = r0 + row;
    const bool rowok = (gr < Nrows);

    // ================= phase 1: LN(x) -> fusion A (f16) =================
    {
        const float4* xp = (const float4*)(x + (size_t)gr * 1024 + sub * 256);
        float s = 0.f, ss = 0.f;
        if (rowok) {
            for (int j = 0; j < 64; ++j) {
                float4 v = xp[j];
                s  += v.x + v.y + v.z + v.w;
                ss += v.x * v.x + v.y * v.y + v.z * v.z + v.w * v.w;
            }
        }
        sRed[row * 8 + sub * 2 + 0] = s;
        sRed[row * 8 + sub * 2 + 1] = ss;
        __syncthreads();
        float S  = sRed[row*8+0] + sRed[row*8+2] + sRed[row*8+4] + sRed[row*8+6];
        float SS = sRed[row*8+1] + sRed[row*8+3] + sRed[row*8+5] + sRed[row*8+7];
        float m    = S * (1.0f / 1024.0f);
        float rstd = rsqrtf(SS * (1.0f / 1024.0f) - m * m + 1e-5f);
        const float4* gg = (const float4*)(en_g + sub * 256);
        const float4* bb = (const float4*)(en_b + sub * 256);
        for (int j = 0; j < 64; ++j) {
            float4 v = rowok ? xp[j] : make_float4(0.f, 0.f, 0.f, 0.f);
            float4 g4 = gg[j], b4 = bb[j];
            int k = row * 1096 + sub * 256 + j * 4;
            sA[k + 0] = (_Float16)((v.x - m) * rstd * g4.x + b4.x);
            sA[k + 1] = (_Float16)((v.y - m) * rstd * g4.y + b4.y);
            sA[k + 2] = (_Float16)((v.z - m) * rstd * g4.z + b4.z);
            sA[k + 3] = (_Float16)((v.w - m) * rstd * g4.w + b4.w);
        }
        __syncthreads();
    }

    // ================= phase 2: context projector (VALU, tiny) =================
    {
        const float4* cp4 = (const float4*)(ctx + (size_t)gr * 256 + sub * 64);
        float s = 0.f, ss = 0.f;
        if (rowok) {
            for (int j = 0; j < 16; ++j) {
                float4 v = cp4[j];
                s  += v.x + v.y + v.z + v.w;
                ss += v.x * v.x + v.y * v.y + v.z * v.z + v.w * v.w;
            }
        }
        sRed[row * 8 + sub * 2 + 0] = s;
        sRed[row * 8 + sub * 2 + 1] = ss;
        __syncthreads();
        float S  = sRed[row*8+0] + sRed[row*8+2] + sRed[row*8+4] + sRed[row*8+6];
        float SS = sRed[row*8+1] + sRed[row*8+3] + sRed[row*8+5] + sRed[row*8+7];
        float m    = S * (1.0f / 256.0f);
        float rstd = rsqrtf(SS * (1.0f / 256.0f) - m * m + 1e-5f);

        // layer1: this thread computes 8 outputs over all 256 inputs
        const int o0 = sub * 8;
        float acc[8];
        for (int o = 0; o < 8; ++o) acc[o] = cp_b1[o0 + o];
        const float4* call = (const float4*)(ctx + (size_t)gr * 256);
        const float4* cg4  = (const float4*)cn_g;
        const float4* cb4  = (const float4*)cn_b;
        const float4* w14  = (const float4*)cp_w1;
        for (int j = 0; j < 64; ++j) {
            float4 v  = rowok ? call[j] : make_float4(0.f, 0.f, 0.f, 0.f);
            float4 g4 = cg4[j], b4 = cb4[j];
            float n0 = (v.x - m) * rstd * g4.x + b4.x;
            float n1 = (v.y - m) * rstd * g4.y + b4.y;
            float n2 = (v.z - m) * rstd * g4.z + b4.z;
            float n3 = (v.w - m) * rstd * g4.w + b4.w;
            #pragma unroll
            for (int o = 0; o < 8; ++o) {
                float4 w = w14[(o0 + o) * 64 + j];
                acc[o] += n0 * w.x + n1 * w.y + n2 * w.z + n3 * w.w;
            }
        }
        __syncthreads();   // all sRed reads complete
        for (int o = 0; o < 8; ++o) sH[row * 548 + o0 + o] = acc[o];
        __syncthreads();

        if (sub == 0) {  // finish projector serially per row (32-dim, tiny)
            float h[32], t[32];
            for (int i = 0; i < 32; ++i) h[i] = sH[row * 548 + i];
            float mm = 0, vv = 0, rs;
            for (int i = 0; i < 32; ++i) mm += h[i];
            mm *= (1.f / 32.f);
            for (int i = 0; i < 32; ++i) { float d = h[i] - mm; vv += d * d; }
            rs = rsqrtf(vv * (1.f / 32.f) + 1e-5f);
            for (int i = 0; i < 32; ++i)
                t[i] = gelu_exact((h[i] - mm) * rs * cp_ln1_g[i] + cp_ln1_b[i]);
            for (int o = 0; o < 32; ++o) {
                float a = cp_b2[o];
                for (int i = 0; i < 32; ++i) a += t[i] * cp_w2[o * 32 + i];
                h[o] = a;
            }
            mm = 0; vv = 0;
            for (int i = 0; i < 32; ++i) mm += h[i];
            mm *= (1.f / 32.f);
            for (int i = 0; i < 32; ++i) { float d = h[i] - mm; vv += d * d; }
            rs = rsqrtf(vv * (1.f / 32.f) + 1e-5f);
            for (int i = 0; i < 32; ++i)
                t[i] = gelu_exact((h[i] - mm) * rs * cp_ln2_g[i] + cp_ln2_b[i]);
            mm = 0; vv = 0;
            for (int i = 0; i < 32; ++i) mm += t[i];
            mm *= (1.f / 32.f);
            for (int i = 0; i < 32; ++i) { float d = t[i] - mm; vv += d * d; }
            rs = rsqrtf(vv * (1.f / 32.f) + 1e-5f);
            for (int i = 0; i < 32; ++i)
                sA[row * 1096 + 1024 + i] = (_Float16)((t[i] - mm) * rs * cf_g[i] + cf_b[i]);
        }
        __syncthreads();
    }

    // ================= GEMM1: [64x1056] x [1056x528] (WMMA f16) =================
    {
        const float8v vz = {0.f, 0.f, 0.f, 0.f, 0.f, 0.f, 0.f, 0.f};
        for (int nt = wv; nt < 33; nt += 8) {
            float8v acc[4];
            #pragma unroll
            for (int m = 0; m < 4; ++m) acc[m] = vz;
            const int n = nt * 16 + lrow;
            const _Float16* bp = Wh1 + (size_t)n * 1056 + lhi * 8;
            for (int ks = 0; ks < 33; ++ks) {
                V16 b;
                b.h[0] = *(const half8v*)(bp + ks * 32);
                b.h[1] = *(const half8v*)(bp + ks * 32 + 16);
                #pragma unroll
                for (int m = 0; m < 4; ++m) {
                    const int abase = (m * 16 + lrow) * 1096 + ks * 32 + lhi * 8;
                    V16 a;
                    a.h[0] = *(const half8v*)(sA + abase);
                    a.h[1] = *(const half8v*)(sA + abase + 16);
                    acc[m] = __builtin_amdgcn_wmma_f32_16x16x32_f16(
                        false, a.v, false, b.v, (short)0, acc[m], false, false);
                }
            }
            const float bias = gp_b1[n];
            #pragma unroll
            for (int m = 0; m < 4; ++m)
                #pragma unroll
                for (int r = 0; r < 8; ++r)
                    sH[(m * 16 + lhi * 8 + r) * 548 + n] = acc[m][r] + bias;
        }
        __syncthreads();
    }

    // ================= LN + GELU (h1) -> A2 (f16, K padded to 544) =================
    {
        const int k0 = sub * 132;
        float s = 0.f, ss = 0.f;
        for (int k = 0; k < 132; ++k) {
            float v = sH[row * 548 + k0 + k];
            s += v; ss += v * v;
        }
        sRed[row * 8 + sub * 2 + 0] = s;
        sRed[row * 8 + sub * 2 + 1] = ss;
        __syncthreads();
        float S  = sRed[row*8+0] + sRed[row*8+2] + sRed[row*8+4] + sRed[row*8+6];
        float SS = sRed[row*8+1] + sRed[row*8+3] + sRed[row*8+5] + sRed[row*8+7];
        float m    = S * (1.0f / 528.0f);
        float rstd = rsqrtf(SS * (1.0f / 528.0f) - m * m + 1e-5f);
        for (int k = 0; k < 132; ++k) {
            int kk = k0 + k;
            float v = sH[row * 548 + kk];
            sA[row * 552 + kk] = (_Float16)gelu_exact((v - m) * rstd * g1g[kk] + g1b[kk]);
        }
        if (sub == 3)
            for (int kk = 528; kk < 544; ++kk) sA[row * 552 + kk] = (_Float16)0.f;
        __syncthreads();
    }

    // ================= GEMM2: [64x544] x [544x272] (WMMA f16) =================
    {
        const float8v vz = {0.f, 0.f, 0.f, 0.f, 0.f, 0.f, 0.f, 0.f};
        for (int nt = wv; nt < 17; nt += 8) {
            float8v acc[4];
            #pragma unroll
            for (int m = 0; m < 4; ++m) acc[m] = vz;
            const int n = nt * 16 + lrow;
            const _Float16* bp = Wh2 + (size_t)n * 544 + lhi * 8;
            for (int ks = 0; ks < 17; ++ks) {
                V16 b;
                b.h[0] = *(const half8v*)(bp + ks * 32);
                b.h[1] = *(const half8v*)(bp + ks * 32 + 16);
                #pragma unroll
                for (int m = 0; m < 4; ++m) {
                    const int abase = (m * 16 + lrow) * 552 + ks * 32 + lhi * 8;
                    V16 a;
                    a.h[0] = *(const half8v*)(sA + abase);
                    a.h[1] = *(const half8v*)(sA + abase + 16);
                    acc[m] = __builtin_amdgcn_wmma_f32_16x16x32_f16(
                        false, a.v, false, b.v, (short)0, acc[m], false, false);
                }
            }
            if (n < 264) {
                const float bias = gp_b2[n];
                #pragma unroll
                for (int m = 0; m < 4; ++m)
                    #pragma unroll
                    for (int r = 0; r < 8; ++r)
                        sH[(m * 16 + lhi * 8 + r) * 276 + n] = acc[m][r] + bias;
            }
        }
        __syncthreads();
    }

    // ================= LN + GELU (h2) -> A3 (f16, K padded to 288) =================
    {
        const int k0 = sub * 66;
        float s = 0.f, ss = 0.f;
        for (int k = 0; k < 66; ++k) {
            float v = sH[row * 276 + k0 + k];
            s += v; ss += v * v;
        }
        sRed[row * 8 + sub * 2 + 0] = s;
        sRed[row * 8 + sub * 2 + 1] = ss;
        __syncthreads();
        float S  = sRed[row*8+0] + sRed[row*8+2] + sRed[row*8+4] + sRed[row*8+6];
        float SS = sRed[row*8+1] + sRed[row*8+3] + sRed[row*8+5] + sRed[row*8+7];
        float m    = S * (1.0f / 264.0f);
        float rstd = rsqrtf(SS * (1.0f / 264.0f) - m * m + 1e-5f);
        for (int k = 0; k < 66; ++k) {
            int kk = k0 + k;
            float v = sH[row * 276 + kk];
            sA[row * 296 + kk] = (_Float16)gelu_exact((v - m) * rstd * g2g[kk] + g2b[kk]);
        }
        if (sub == 3)
            for (int kk = 264; kk < 288; ++kk) sA[row * 296 + kk] = (_Float16)0.f;
        __syncthreads();
    }

    // ================= GEMM3: [64x288] x [288x16] -> logits (WMMA f16) =================
    {
        if (wv < 4) {   // wave-uniform: EXEC all-ones inside
            const float8v vz = {0.f, 0.f, 0.f, 0.f, 0.f, 0.f, 0.f, 0.f};
            float8v acc = vz;
            const int mt = wv;
            const _Float16* bp = Wh3 + (size_t)lrow * 288 + lhi * 8;
            for (int ks = 0; ks < 9; ++ks) {
                V16 b;
                b.h[0] = *(const half8v*)(bp + ks * 32);
                b.h[1] = *(const half8v*)(bp + ks * 32 + 16);
                const int abase = (mt * 16 + lrow) * 296 + ks * 32 + lhi * 8;
                V16 a;
                a.h[0] = *(const half8v*)(sA + abase);
                a.h[1] = *(const half8v*)(sA + abase + 16);
                acc = __builtin_amdgcn_wmma_f32_16x16x32_f16(
                    false, a.v, false, b.v, (short)0, acc, false, false);
            }
            const float bias = gp_b3[lrow];
            #pragma unroll
            for (int r = 0; r < 8; ++r)
                sLog[(mt * 16 + lhi * 8 + r) * 16 + lrow] = acc[r] + bias;
        }
        __syncthreads();
    }

    // ================= top-2 + softmax + outputs =================
    if (tid < 64 && (r0 + tid) < Nrows) {
        const int rr = tid;
        float l[16];
        for (int i = 0; i < 16; ++i) l[i] = sLog[rr * 16 + i];
        int i0 = 0; float m0 = l[0];
        for (int i = 1; i < 16; ++i) if (l[i] > m0) { m0 = l[i]; i0 = i; }
        int i1 = (i0 == 0) ? 1 : 0; float m1 = l[i1];
        for (int i = 0; i < 16; ++i)
            if (i != i0 && l[i] > m1) { m1 = l[i]; i1 = i; }
        float e  = expf(m1 - m0);
        float w0 = 1.0f / (1.0f + e);
        float w1 = e / (1.0f + e);
        const size_t gRow = (size_t)r0 + rr;
        out[gRow * 2 + 0] = w0;
        out[gRow * 2 + 1] = w1;
        float* oidx = out + (size_t)2 * Nrows;
        oidx[gRow * 2 + 0] = (float)i0;
        oidx[gRow * 2 + 1] = (float)i1;
        float* olog = out + (size_t)4 * Nrows;
        for (int i = 0; i < 16; ++i) olog[gRow * 16 + i] = l[i];
    }
}

extern "C" void kernel_launch(void* const* d_in, const int* in_sizes, int n_in,
                              void* d_out, int out_size, void* d_ws, size_t ws_size,
                              hipStream_t stream) {
    // input order per setup_inputs()
    const float* x        = (const float*)d_in[0];
    const float* context  = (const float*)d_in[1];
    const float* en_g     = (const float*)d_in[2];
    const float* en_b     = (const float*)d_in[3];
    const float* cn_g     = (const float*)d_in[4];
    const float* cn_b     = (const float*)d_in[5];
    const float* cp_w1    = (const float*)d_in[6];
    const float* cp_b1    = (const float*)d_in[7];
    const float* cp_ln1_g = (const float*)d_in[8];
    const float* cp_ln1_b = (const float*)d_in[9];
    const float* cp_w2    = (const float*)d_in[10];
    const float* cp_b2    = (const float*)d_in[11];
    const float* cp_ln2_g = (const float*)d_in[12];
    const float* cp_ln2_b = (const float*)d_in[13];
    const float* cf_g     = (const float*)d_in[14];
    const float* cf_b     = (const float*)d_in[15];
    const float* gp_w1    = (const float*)d_in[16];
    const float* gp_b1    = (const float*)d_in[17];
    const float* g1g      = (const float*)d_in[18];
    const float* g1b      = (const float*)d_in[19];
    const float* gp_w2    = (const float*)d_in[20];
    const float* gp_b2    = (const float*)d_in[21];
    const float* g2g      = (const float*)d_in[22];
    const float* g2b      = (const float*)d_in[23];
    const float* gp_w3    = (const float*)d_in[24];
    const float* gp_b3    = (const float*)d_in[25];

    const int Nrows = in_sizes[0] / 1024;

    _Float16* wsH = (_Float16*)d_ws;   // ~1.4 MB of packed f16 weights
    const int W1E = 528 * 1056;
    const int W2E = 272 * 544;
    const int W3E = 16 * 288;
    const int WTOT = W1E + W2E + W3E;

    pack_weights_kernel<<<(WTOT + 255) / 256, 256, 0, stream>>>(gp_w1, gp_w2, gp_w3, wsH);

    dim3 grid((Nrows + 63) / 64);
    gating_fused_kernel<<<grid, 256, 0, stream>>>(
        x, context, en_g, en_b, cn_g, cn_b,
        cp_w1, cp_b1, cp_ln1_g, cp_ln1_b, cp_w2, cp_b2, cp_ln2_g, cp_ln2_b,
        cf_g, cf_b,
        gp_b1, g1g, g1b, gp_b2, g2g, g2b, gp_b3,
        wsH, wsH + W1E, wsH + W1E + W2E,
        (float*)d_out, Nrows);
}
